// Gru_cond_layer_aam_71098888618167
// MI455X (gfx1250) — compile-verified, matrix operand order checked by reading, not run on hardware
//
#include <hip/hip_runtime.h>
#include <math.h>

// ---------------- problem constants ----------------
#define T_   32
#define B_   16
#define M_   256
#define N_   256
#define D_   684
#define A_   512
#define H_   16
#define W_   32
#define HW_  512
#define QC_  512
#define DK_  704   // D padded to multiple of 32
#define KF_  160   // fused conv-tap K (121 + 25 -> pad 160)

// ---------------- WMMA helpers (CDNA5 gfx1250, wave32) ----------------
typedef __attribute__((ext_vector_type(16))) __bf16 bf16x16;
typedef __attribute__((ext_vector_type(8)))  float  f32x8;

// K index held by VGPR v (0..7) for lane-half `half` per the 16-bit A/B layout
// (ISA 7.12.2): VGPR0..3 -> K = 2v (+8 for upper half), VGPR4..7 -> +16.
__device__ __forceinline__ int wk(int v, int half) {
  return ((v & 4) << 2) + ((v & 3) << 1) + (half << 3);
}

// Scalar-gather fragment builder (used only in one-time precompute kernels).
template <typename F>
__device__ __forceinline__ bf16x16 mk_frag(F&& get, int k0, int lane) {
  const int r = lane & 15, h = lane >> 4;
  bf16x16 f;
#pragma unroll
  for (int v = 0; v < 8; ++v) {
    const int k = k0 + wk(v, h);
    f[2 * v]     = (__bf16)get(r, k);
    f[2 * v + 1] = (__bf16)get(r, k + 1);
  }
  return f;
}

// Load a pre-packed fragment: 32 contiguous bytes per lane.
__device__ __forceinline__ bf16x16 ld_frag(const __bf16* base, int tile, int lane) {
  return ((const bf16x16*)base)[tile * 32 + lane];
}

// Pack an A-operand into fragment-order bf16 in LDS (whole workgroup).
// get(r,k) supplies the f32 element; Kpad must be a multiple of 32.
template <typename F>
__device__ __forceinline__ void pack_A_lds(__bf16* Ap, F&& get, int Kpad,
                                           int tid, int nthreads) {
  const int total = (Kpad >> 5) * 512;   // ktiles * 32 lanes * 16 elems
  for (int idx = tid; idx < total; idx += nthreads) {
    const int e = idx & 15, lane = (idx >> 4) & 31, kt = idx >> 9;
    const int r = lane & 15, h = lane >> 4;
    const int k = (kt << 5) + wk(e >> 1, h) + (e & 1);
    Ap[idx] = (__bf16)get(r, k);
  }
}

__device__ __forceinline__ f32x8 wmma_bf16(bf16x16 a, bf16x16 b, f32x8 c) {
  return __builtin_amdgcn_wmma_f32_16x16x32_bf16(false, a, false, b,
                                                 (short)0, c, false, false);
}

__device__ __forceinline__ float sigmoidf_(float x) {
  return 1.0f / (1.0f + __expf(-x));
}

// ---------------- weight pre-packing ----------------
// Pack W (Nrows x K row-major) into frag-order bf16, K padded to Kpad.
__global__ void k_pack(const float* __restrict__ W, int Nrows, int K, int Kpad,
                       __bf16* __restrict__ out) {
  const int idx = blockIdx.x * blockDim.x + threadIdx.x;
  const int KT = Kpad >> 5;
  const int total = (Nrows >> 4) * KT * 512;
  if (idx >= total) return;
  const int e = idx & 15, lane = (idx >> 4) & 31;
  const int kt = (idx >> 9) % KT, nt = (idx >> 9) / KT;
  const int n = lane & 15, h = lane >> 4;
  const int k = (kt << 5) + wk(e >> 1, h) + (e & 1);
  const float v = (k < K) ? W[(size_t)(nt * 16 + n) * K + k] : 0.0f;
  out[idx] = (__bf16)v;
}

// ---------------- one-time precompute kernels ----------------

// sb{z,r,h} = embedding(512x256) @ Wy*_w^T + bias (runs once; scalar-gather ok)
__global__ void k_sb(const float* __restrict__ emb,
                     const float* __restrict__ Wz, const float* __restrict__ bz,
                     const float* __restrict__ Wr, const float* __restrict__ br,
                     const float* __restrict__ Wh, const float* __restrict__ bh,
                     float* __restrict__ sbz, float* __restrict__ sbr,
                     float* __restrict__ sbh) {
  const int nw   = blockDim.x >> 5;
  const int wid  = blockIdx.x * nw + (threadIdx.x >> 5);
  const int lane = threadIdx.x & 31;
  const int MT = (T_ * B_) / 16, NT = N_ / 16;
  if (wid >= 3 * MT * NT) return;                        // wave-uniform
  const int g = wid / (MT * NT), rem = wid % (MT * NT);
  const int mt = rem / NT, nt = rem % NT;
  const float* Wg = (g == 0) ? Wz : (g == 1) ? Wr : Wh;
  const float* bg = (g == 0) ? bz : (g == 1) ? br : bh;
  float* dst = (g == 0) ? sbz : (g == 1) ? sbr : sbh;
  const float* Arow = emb + (size_t)(mt * 16) * M_;
  f32x8 acc = {};
  for (int k0 = 0; k0 < M_; k0 += 32) {
    bf16x16 a = mk_frag([&](int r, int k) { return Arow[r * M_ + k]; }, k0, lane);
    bf16x16 b = mk_frag([&](int n, int k) { return Wg[(nt * 16 + n) * M_ + k]; }, k0, lane);
    acc = wmma_bf16(a, b, acc);
  }
  const int n = lane & 15, mb = (lane >> 4) * 8;
#pragma unroll
  for (int i = 0; i < 8; ++i)
    dst[(size_t)(mt * 16 + mb + i) * N_ + nt * 16 + n] = acc[i] + bg[nt * 16 + n];
}

// Ua_ctx[hw][b][a] = sum_d context[b,d,hw] * Ua_w[a,d] + Ua_b[a] (runs once)
__global__ void k_uactx(const float* __restrict__ ctx, const float* __restrict__ Uaw,
                        const float* __restrict__ Uab, float* __restrict__ Ua_ctx) {
  const int nw   = blockDim.x >> 5;
  const int wid  = blockIdx.x * nw + (threadIdx.x >> 5);
  const int lane = threadIdx.x & 31;
  const int NT = A_ / 16;
  if (wid >= HW_ * NT) return;                           // wave-uniform
  const int hw = wid / NT, nt = wid % NT;
  f32x8 acc = {};
  for (int k0 = 0; k0 < DK_; k0 += 32) {
    bf16x16 a = mk_frag(
        [&](int b, int k) { return (k < D_) ? ctx[((size_t)b * D_ + k) * HW_ + hw] : 0.0f; },
        k0, lane);
    bf16x16 bb = mk_frag(
        [&](int n, int k) { return (k < D_) ? Uaw[(size_t)(nt * 16 + n) * D_ + k] : 0.0f; },
        k0, lane);
    acc = wmma_bf16(a, bb, acc);
  }
  const int n = lane & 15, mb = (lane >> 4) * 8;
#pragma unroll
  for (int i = 0; i < 8; ++i)
    Ua_ctx[((size_t)hw * B_ + mb + i) * A_ + nt * 16 + n] = acc[i] + Uab[nt * 16 + n];
}

// Fuse conv filters with Uf projections into a 512 x 160 tap matrix.
__global__ void k_fcat(const float* __restrict__ Uf, const float* __restrict__ Uf2,
                       const float* __restrict__ Qw, const float* __restrict__ Q2w,
                       const float* __restrict__ Ufb, const float* __restrict__ Uf2b,
                       float* __restrict__ Fcat, float* __restrict__ fb) {
  const int idx = blockIdx.x * blockDim.x + threadIdx.x;
  if (idx >= A_ * KF_) return;
  const int a = idx / KF_, j = idx % KF_;
  float s = 0.0f;
  if (j < 121) {
    for (int q = 0; q < QC_; ++q) s += Uf[a * QC_ + q] * Qw[q * 121 + j];
  } else if (j < 146) {
    const int jj = j - 121;
    for (int q = 0; q < QC_; ++q) s += Uf2[a * QC_ + q] * Q2w[q * 25 + jj];
  }
  Fcat[a * KF_ + j] = s;
  if (j == 0) fb[a] = Ufb[a] + Uf2b[a];
}

__global__ void k_init(const float* __restrict__ h0, float* __restrict__ h_cur,
                       float* __restrict__ ap) {
  const int i = blockIdx.x * blockDim.x + threadIdx.x;
  if (i < B_ * N_) h_cur[i] = h0[i];
  if (i < B_ * HW_) ap[i] = 0.0f;
}

// ---------------- per-timestep kernels ----------------

// GRU layer 1 + Wa projection. Single workgroup (M=16 is one tile row).
__global__ void __launch_bounds__(256) k_gru1(
    const float* __restrict__ h_cur, const float* __restrict__ sbz,
    const float* __restrict__ sbr, const float* __restrict__ sbh,
    const __bf16* __restrict__ Uz_p, const __bf16* __restrict__ Ur_p,
    const __bf16* __restrict__ Uh_p, const __bf16* __restrict__ Wa_p,
    const float* __restrict__ mask, int t,
    float* __restrict__ h1_out, float* __restrict__ Wa_h1) {
  __shared__ float g[3][B_ * N_];       // z,r,h pre-activations; g[0] later = h1
  __shared__ bf16x16 Ah[8 * 32];        // A operand, frag order (K = 256)
  const int tid = threadIdx.x, lane = tid & 31, wave = tid >> 5, nw = blockDim.x >> 5;

  pack_A_lds((__bf16*)Ah, [&](int r, int k) { return h_cur[r * N_ + k]; },
             N_, tid, blockDim.x);
  __syncthreads();

  for (int tile = wave; tile < 48; tile += nw) {
    const int gi = tile / 16, nt = tile % 16;
    const __bf16* Up = (gi == 0) ? Uz_p : (gi == 1) ? Ur_p : Uh_p;
    f32x8 acc = {};
#pragma unroll
    for (int kt = 0; kt < 8; ++kt)
      acc = wmma_bf16(Ah[kt * 32 + lane], ld_frag(Up, nt * 8 + kt, lane), acc);
    const int n = lane & 15, mb = (lane >> 4) * 8;
#pragma unroll
    for (int i = 0; i < 8; ++i) g[gi][(mb + i) * N_ + nt * 16 + n] = acc[i];
  }
  __syncthreads();

  const float* sz = sbz + (size_t)t * B_ * N_;
  const float* sr = sbr + (size_t)t * B_ * N_;
  const float* sh = sbh + (size_t)t * B_ * N_;
  for (int i = tid; i < B_ * N_; i += blockDim.x) {
    const int b = i / N_;
    const float m = mask[t * B_ + b];
    const float hp = h_cur[i];
    const float z1  = sigmoidf_(g[0][i] + sz[i]);
    const float r1  = sigmoidf_(g[1][i] + sr[i]);
    const float h1p = tanhf(g[2][i] * r1 + sh[i]);
    float h1 = z1 * hp + (1.0f - z1) * h1p;
    h1 = m * h1 + (1.0f - m) * hp;
    g[0][i] = h1;
    h1_out[i] = h1;
  }
  __syncthreads();
  pack_A_lds((__bf16*)Ah, [&](int r, int k) { return g[0][r * N_ + k]; },
             N_, tid, blockDim.x);
  __syncthreads();

  // Wa_h1 = h1 @ Wa^T : 16x512, 32 column tiles
  for (int nt = wave; nt < A_ / 16; nt += nw) {
    f32x8 acc = {};
#pragma unroll
    for (int kt = 0; kt < 8; ++kt)
      acc = wmma_bf16(Ah[kt * 32 + lane], ld_frag(Wa_p, nt * 8 + kt, lane), acc);
    const int n = lane & 15, mb = (lane >> 4) * 8;
#pragma unroll
    for (int i = 0; i < 8; ++i) Wa_h1[(mb + i) * A_ + nt * 16 + n] = acc[i];
  }
}

// Fused coverage-conv (GEMM over 146 shifted taps) + score tanh + v_a reduce.
// One workgroup per spatial position hw; M tile = all 16 batches.
__global__ void __launch_bounds__(256) k_score(
    const float* __restrict__ ap, const __bf16* __restrict__ Fp,
    const float* __restrict__ fb, const float* __restrict__ Ua_ctx,
    const float* __restrict__ Wa_h1, const float* __restrict__ va_w,
    float* __restrict__ e) {
  __shared__ bf16x16 Ap[5 * 32];        // patch, frag order (K = 160)
  __shared__ float epart[B_];
  const int hw = blockIdx.x, h = hw / W_, w = hw % W_;
  const int tid = threadIdx.x;

  // gather zero-padded 11x11 + 5x5 neighborhoods directly into frag order
  pack_A_lds((__bf16*)Ap,
             [&](int b, int k) {
               float v = 0.0f;
               if (k < 121) {
                 const int y = h + k / 11 - 5, x = w + k % 11 - 5;
                 if ((unsigned)y < H_ && (unsigned)x < W_) v = ap[b * HW_ + y * W_ + x];
               } else if (k < 146) {
                 const int jj = k - 121;
                 const int y = h + jj / 5 - 2, x = w + jj % 5 - 2;
                 if ((unsigned)y < H_ && (unsigned)x < W_) v = ap[b * HW_ + y * W_ + x];
               }
               return v;
             },
             KF_, tid, blockDim.x);
  if (tid < B_) epart[tid] = 0.0f;
  __syncthreads();

  const int lane = tid & 31, wave = tid >> 5, nw = blockDim.x >> 5;
  const float* Ua = Ua_ctx + (size_t)hw * B_ * A_;
  __builtin_prefetch(Ua, 0, 0);                 // global_prefetch_b8

  for (int nt = wave; nt < A_ / 16; nt += nw) {
    f32x8 acc = {};
#pragma unroll
    for (int kt = 0; kt < 5; ++kt)
      acc = wmma_bf16(Ap[kt * 32 + lane], ld_frag(Fp, nt * 5 + kt, lane), acc);
    const int n = lane & 15, mb = (lane >> 4) * 8;
    const int col = nt * 16 + n;
    const float va = va_w[col], fbv = fb[col];
#pragma unroll
    for (int i = 0; i < 8; ++i) {
      const int b = mb + i;
      const float sc = tanhf(acc[i] + Ua[b * A_ + col] + Wa_h1[b * A_ + col] + fbv);
      atomicAdd(&epart[b], sc * va);            // ds_add_f32 reduction over A
    }
  }
  __syncthreads();
  if (tid < B_) e[hw * B_ + tid] = epart[tid];
}

// softmax over (h,w) per batch + alpha_past update + outputs
__global__ void __launch_bounds__(256) k_softmax(
    const float* __restrict__ e, const float* __restrict__ va_b,
    const float* __restrict__ cmask, float* __restrict__ ap,
    float* __restrict__ alpha_cur, float* __restrict__ out_alphas,
    float* __restrict__ out_apast, int t) {
  __shared__ float sred[256];
  const int b = blockIdx.x, tid = threadIdx.x;
  float vals[HW_ / 256];
  float local = 0.0f;
#pragma unroll
  for (int i = 0; i < HW_ / 256; ++i) {
    const int hw = i * 256 + tid;
    const float v = __expf(e[hw * B_ + b] + va_b[0]) * cmask[b * HW_ + hw];
    vals[i] = v;
    local += v;
  }
  sred[tid] = local;
  __syncthreads();
  for (int s = 128; s > 0; s >>= 1) {
    if (tid < s) sred[tid] += sred[tid + s];
    __syncthreads();
  }
  const float inv = 1.0f / sred[0];
#pragma unroll
  for (int i = 0; i < HW_ / 256; ++i) {
    const int hw = i * 256 + tid;
    const float al = vals[i] * inv;
    alpha_cur[b * HW_ + hw] = al;
    out_alphas[(size_t)t * B_ * HW_ + b * HW_ + hw] = al;
    const float apn = ap[b * HW_ + hw] + al;
    ap[b * HW_ + hw] = apn;
    out_apast[(size_t)t * B_ * HW_ + b * HW_ + hw] = apn;
  }
}

// ct[b,d] = sum_hw context[b,d,hw] * alpha[b,hw]   (one wave per row)
__global__ void __launch_bounds__(256) k_ct(
    const float* __restrict__ ctx, const float* __restrict__ alpha,
    float* __restrict__ ct, float* __restrict__ out_ct, int t) {
  const int wave = threadIdx.x >> 5, lane = threadIdx.x & 31;
  const int row = blockIdx.x * (blockDim.x >> 5) + wave;
  if (row >= B_ * D_) return;                    // wave-uniform
  const int b = row / D_;
  const float* crow = ctx + (size_t)row * HW_;
  const float* arow = alpha + b * HW_;
  float s = 0.0f;
  for (int i = lane; i < HW_; i += 32) s += crow[i] * arow[i];
#pragma unroll
  for (int m = 16; m > 0; m >>= 1) s += __shfl_xor(s, m, 32);
  if (lane == 0) {
    ct[row] = s;
    out_ct[(size_t)t * B_ * D_ + row] = s;
  }
}

// GRU layer 2: z/r combine ct- and h1- GEMMs; h-gate keeps the parts split.
__global__ void __launch_bounds__(256) k_gru2(
    const float* __restrict__ ct, const float* __restrict__ h1,
    const __bf16* __restrict__ Wcz_p, const __bf16* __restrict__ Wcr_p,
    const __bf16* __restrict__ Wch_p, const __bf16* __restrict__ Uz2_p,
    const float* __restrict__ bz2, const __bf16* __restrict__ Ur2_p,
    const float* __restrict__ br2, const __bf16* __restrict__ Uh2_p,
    const float* __restrict__ bh2, const float* __restrict__ mask, int t,
    float* __restrict__ h_cur, float* __restrict__ out_h2) {
  __shared__ float G[4][B_ * N_];       // Gz(ct+h1), Gr(ct+h1), Gch(ct), Ghh(h1)
  __shared__ bf16x16 Act[22 * 32];      // ct operand, K = 704
  __shared__ bf16x16 Ah1[8 * 32];       // h1 operand, K = 256
  const int tid = threadIdx.x, lane = tid & 31, wave = tid >> 5, nw = blockDim.x >> 5;

  pack_A_lds((__bf16*)Act,
             [&](int r, int k) { return (k < D_) ? ct[r * D_ + k] : 0.0f; },
             DK_, tid, blockDim.x);
  pack_A_lds((__bf16*)Ah1, [&](int r, int k) { return h1[r * N_ + k]; },
             N_, tid, blockDim.x);
  __syncthreads();

  for (int tile = wave; tile < 64; tile += nw) {
    const int gi = tile >> 4, nt = tile & 15;
    f32x8 acc = {};
    if (gi < 3) {  // ct @ Wc{z,r,h}^T, K = 704
      const __bf16* Wp = (gi == 0) ? Wcz_p : (gi == 1) ? Wcr_p : Wch_p;
#pragma unroll
      for (int kt = 0; kt < 22; ++kt)
        acc = wmma_bf16(Act[kt * 32 + lane], ld_frag(Wp, nt * 22 + kt, lane), acc);
    }
    if (gi != 2) {  // h1 @ Uh{z2,r2,h2}^T, K = 256 (z/r accumulate on top)
      const __bf16* Up = (gi == 0) ? Uz2_p : (gi == 1) ? Ur2_p : Uh2_p;
#pragma unroll
      for (int kt = 0; kt < 8; ++kt)
        acc = wmma_bf16(Ah1[kt * 32 + lane], ld_frag(Up, nt * 8 + kt, lane), acc);
    }
    const int n = lane & 15, mb = (lane >> 4) * 8;
#pragma unroll
    for (int i = 0; i < 8; ++i) G[gi][(mb + i) * N_ + nt * 16 + n] = acc[i];
  }
  __syncthreads();

  for (int i = tid; i < B_ * N_; i += blockDim.x) {
    const int b = i / N_, n = i % N_;
    const float m = mask[t * B_ + b];
    const float h1v = h1[i];
    const float z2  = sigmoidf_(G[0][i] + bz2[n]);
    const float r2  = sigmoidf_(G[1][i] + br2[n]);
    const float h2p = tanhf(G[2][i] + (G[3][i] + bh2[n]) * r2);
    float h2 = z2 * h1v + (1.0f - z2) * h2p;
    h2 = m * h2 + (1.0f - m) * h1v;
    h_cur[i] = h2;
    out_h2[(size_t)t * B_ * N_ + i] = h2;
  }
}

// ---------------- host driver ----------------
extern "C" void kernel_launch(void* const* d_in, const int* in_sizes, int n_in,
                              void* d_out, int out_size, void* d_ws, size_t ws_size,
                              hipStream_t stream) {
  (void)in_sizes; (void)n_in; (void)out_size; (void)ws_size;
  const float* emb    = (const float*)d_in[0];
  const float* mask   = (const float*)d_in[1];
  const float* ctx    = (const float*)d_in[2];
  const float* cmask  = (const float*)d_in[3];
  const float* h0     = (const float*)d_in[4];
  const float* Ua_w   = (const float*)d_in[5];
  const float* Ua_b   = (const float*)d_in[6];
  const float* Wa     = (const float*)d_in[7];
  const float* Q_w    = (const float*)d_in[8];
  const float* Q2_w   = (const float*)d_in[9];
  const float* Uf_w   = (const float*)d_in[10];
  const float* Uf_b   = (const float*)d_in[11];
  const float* Uf2_w  = (const float*)d_in[12];
  const float* Uf2_b  = (const float*)d_in[13];
  const float* va_w   = (const float*)d_in[14];
  const float* va_b   = (const float*)d_in[15];
  const float* Wyz_w  = (const float*)d_in[16];
  const float* Wyz_b  = (const float*)d_in[17];
  const float* Wyr_w  = (const float*)d_in[18];
  const float* Wyr_b  = (const float*)d_in[19];
  const float* Wyh_w  = (const float*)d_in[20];
  const float* Wyh_b  = (const float*)d_in[21];
  const float* Uhz    = (const float*)d_in[22];
  const float* Uhr    = (const float*)d_in[23];
  const float* Uhh    = (const float*)d_in[24];
  const float* Wcz    = (const float*)d_in[25];
  const float* Wcr    = (const float*)d_in[26];
  const float* Wch    = (const float*)d_in[27];
  const float* Uhz2_w = (const float*)d_in[28];
  const float* Uhz2_b = (const float*)d_in[29];
  const float* Uhr2_w = (const float*)d_in[30];
  const float* Uhr2_b = (const float*)d_in[31];
  const float* Uhh2_w = (const float*)d_in[32];
  const float* Uhh2_b = (const float*)d_in[33];

  // outputs (concatenated flat in return order)
  float* out       = (float*)d_out;
  float* out_h2    = out;                                  // T*B*N
  float* out_ct    = out + (size_t)T_ * B_ * N_;           // T*B*D
  float* out_alpha = out_ct + (size_t)T_ * B_ * D_;        // T*B*H*W
  float* out_apast = out_alpha + (size_t)T_ * B_ * HW_;    // T*B*H*W

  // workspace carve-up: f32 region, then 64B-aligned bf16 packed-weight region
  float* ws = (float*)d_ws;
  size_t o = 0;
  float* h_cur  = ws + o; o += B_ * N_;
  float* h1     = ws + o; o += B_ * N_;
  float* Wa_h1  = ws + o; o += B_ * A_;
  float* ap     = ws + o; o += B_ * HW_;
  float* e      = ws + o; o += HW_ * B_;
  float* alpha  = ws + o; o += B_ * HW_;
  float* ctv    = ws + o; o += B_ * D_;
  float* sbz    = ws + o; o += (size_t)T_ * B_ * N_;
  float* sbr    = ws + o; o += (size_t)T_ * B_ * N_;
  float* sbh    = ws + o; o += (size_t)T_ * B_ * N_;
  float* Fcat   = ws + o; o += (size_t)A_ * KF_;
  float* fb     = ws + o; o += A_;
  float* Ua_ctx = ws + o; o += (size_t)HW_ * B_ * A_;

  __bf16* bp = (__bf16*)(((uintptr_t)(ws + o) + 63) & ~(uintptr_t)63);
  size_t bo = 0;
  __bf16* Uz_p  = bp + bo; bo += (size_t)(N_/16) * (N_/32)  * 512;
  __bf16* Ur_p  = bp + bo; bo += (size_t)(N_/16) * (N_/32)  * 512;
  __bf16* Uh_p  = bp + bo; bo += (size_t)(N_/16) * (N_/32)  * 512;
  __bf16* Wa_p  = bp + bo; bo += (size_t)(A_/16) * (N_/32)  * 512;
  __bf16* Fp    = bp + bo; bo += (size_t)(A_/16) * (KF_/32) * 512;
  __bf16* Wcz_p = bp + bo; bo += (size_t)(N_/16) * (DK_/32) * 512;
  __bf16* Wcr_p = bp + bo; bo += (size_t)(N_/16) * (DK_/32) * 512;
  __bf16* Wch_p = bp + bo; bo += (size_t)(N_/16) * (DK_/32) * 512;
  __bf16* Uz2_p = bp + bo; bo += (size_t)(N_/16) * (N_/32)  * 512;
  __bf16* Ur2_p = bp + bo; bo += (size_t)(N_/16) * (N_/32)  * 512;
  __bf16* Uh2_p = bp + bo; bo += (size_t)(N_/16) * (N_/32)  * 512;

  auto pack = [&](const float* W, int Nrows, int K, int Kpad, __bf16* dst) {
    const int total = (Nrows / 16) * (Kpad / 32) * 512;
    k_pack<<<(total + 255) / 256, 256, 0, stream>>>(W, Nrows, K, Kpad, dst);
  };

  // ---- one-time precompute (deterministic; re-runs each call)
  k_init<<<(B_ * HW_ + 255) / 256, 256, 0, stream>>>(h0, h_cur, ap);
  k_sb<<<(3 * 32 * 16) / 8, 256, 0, stream>>>(emb, Wyz_w, Wyz_b, Wyr_w, Wyr_b,
                                              Wyh_w, Wyh_b, sbz, sbr, sbh);
  k_uactx<<<(HW_ * (A_ / 16)) / 8, 256, 0, stream>>>(ctx, Ua_w, Ua_b, Ua_ctx);
  k_fcat<<<(A_ * KF_ + 255) / 256, 256, 0, stream>>>(Uf_w, Uf2_w, Q_w, Q2_w,
                                                     Uf_b, Uf2_b, Fcat, fb);
  pack(Uhz,    N_, N_,  N_,  Uz_p);
  pack(Uhr,    N_, N_,  N_,  Ur_p);
  pack(Uhh,    N_, N_,  N_,  Uh_p);
  pack(Wa,     A_, N_,  N_,  Wa_p);
  pack(Fcat,   A_, KF_, KF_, Fp);
  pack(Wcz,    N_, D_,  DK_, Wcz_p);
  pack(Wcr,    N_, D_,  DK_, Wcr_p);
  pack(Wch,    N_, D_,  DK_, Wch_p);
  pack(Uhz2_w, N_, N_,  N_,  Uz2_p);
  pack(Uhr2_w, N_, N_,  N_,  Ur2_p);
  pack(Uhh2_w, N_, N_,  N_,  Uh2_p);

  // ---- sequential scan over T timesteps
  for (int t = 0; t < T_; ++t) {
    k_gru1<<<1, 256, 0, stream>>>(h_cur, sbz, sbr, sbh, Uz_p, Ur_p, Uh_p, Wa_p,
                                  mask, t, h1, Wa_h1);
    k_score<<<HW_, 256, 0, stream>>>(ap, Fp, fb, Ua_ctx, Wa_h1, va_w, e);
    k_softmax<<<B_, 256, 0, stream>>>(e, va_b, cmask, ap, alpha, out_alpha,
                                      out_apast, t);
    k_ct<<<(B_ * D_ + 7) / 8, 256, 0, stream>>>(ctx, alpha, ctv, out_ct, t);
    k_gru2<<<1, 256, 0, stream>>>(ctv, h1, Wcz_p, Wcr_p, Wch_p, Uz2_p, Uhz2_b,
                                  Ur2_p, Uhr2_b, Uh2_p, Uhh2_b, mask, t,
                                  h_cur, out_h2);
  }
}
